// syl_model_82592221102517
// MI455X (gfx1250) — matrix operand order repaired and checked by reading, not compiled, not run on hardware
//
#include <hip/hip_runtime.h>
#include <hip/hip_bf16.h>

// ---- problem constants ----
#define B_    4096
#define L_    30
#define W_    5
#define A_    30
#define E_    128
#define C_    40
#define H_    128
#define T_    3
#define LIN_  200
#define G4_   512            // 4*H
#define N_    (B_*L_)        // 122880
#define KTF_  7              // ceil(224/32): feature K padded 200 -> 224
#define KTH_  4              // 128/32
#define NT_   32             // 512/16 output column tiles
#define BT_   256            // 4096/16 batch row tiles
#define EPS_  1e-5f

typedef __attribute__((ext_vector_type(16))) _Float16 v16h;
typedef __attribute__((ext_vector_type(8)))  float    v8f;

// A-matrix (16x32 f16) per-lane element mapping: half = lane>>4, element j -> K
//   j<8 : K = 8*half + j ;  j>=8 : K = 16 + 8*half + (j-8)
// inverse (ko -> half,j): half = (ko>>3)&1 ; j = (ko&7) + ((ko&16)>>1)
__device__ __forceinline__ void a_swizzle(int ko, int& half, int& j) {
  half = (ko >> 3) & 1;
  j = (ko & 7) + ((ko & 16) >> 1);
}

// ---------------------------------------------------------------------------
// Kernel 1: fold embedding * conv * BN1 into a tiny table
//   Mfold[k][a][c] = bn1scale[c] * sum_e conv_w[c,e,k]*emb[a,e]
//   biasf[c]       = (conv_b[c]-bn1_m[c])*bn1scale[c] + bn1_b[c]
// ---------------------------------------------------------------------------
__global__ void build_tables(const float* __restrict__ emb,
                             const float* __restrict__ conv_w,
                             const float* __restrict__ conv_b,
                             const float* __restrict__ g1, const float* __restrict__ b1,
                             const float* __restrict__ m1, const float* __restrict__ v1,
                             float* __restrict__ Mfold, float* __restrict__ biasf) {
  int idx = blockIdx.x * blockDim.x + threadIdx.x;
  if (idx < 3 * A_ * C_) {
    int c = idx % C_;
    int a = (idx / C_) % A_;
    int k = idx / (C_ * A_);
    float s = g1[c] * rsqrtf(v1[c] + EPS_);
    float d = 0.f;
    for (int e = 0; e < E_; ++e) d += conv_w[(c * E_ + e) * 3 + k] * emb[a * E_ + e];
    Mfold[(k * A_ + a) * C_ + c] = d * s;
  }
  if (idx < C_) {
    float s = g1[idx] * rsqrtf(v1[idx] + EPS_);
    biasf[idx] = (conv_b[idx] - m1[idx]) * s + b1[idx];
  }
}

// ---------------------------------------------------------------------------
// Kernel 2: pack w_ih (both dirs) into WMMA B-layout, K padded 200 -> 224
//   layout [dir][kt][nt][lane][16] ; lane%16 = column n (gate), K swizzled as A-layout/lane
// ---------------------------------------------------------------------------
__global__ void pack_wih(const float* __restrict__ wf, const float* __restrict__ wb,
                         _Float16* __restrict__ out) {
  int idx = blockIdx.x * blockDim.x + threadIdx.x;      // 2*7*32*32*16 = 229376
  int j    = idx & 15;
  int lane = (idx >> 4) & 31;
  int nt   = (idx >> 9) & (NT_ - 1);
  int kt   = (idx >> 14) % KTF_;
  int dir  = idx / (16 * 32 * NT_ * KTF_);
  int k = kt * 32 + ((lane < 16) ? 0 : 8) + (j & 7) + ((j & 8) << 1);
  int g = nt * 16 + (lane & 15);
  const float* w = dir ? wb : wf;
  float val = (k < LIN_) ? w[g * LIN_ + k] : 0.f;
  out[idx] = (_Float16)val;
}

// Kernel 3: pack w_hh (both dirs) into B-layout, K = 128 (no pad)
__global__ void pack_whh(const float* __restrict__ wf, const float* __restrict__ wb,
                         _Float16* __restrict__ out) {
  int idx = blockIdx.x * blockDim.x + threadIdx.x;      // 2*4*32*32*16 = 131072
  int j    = idx & 15;
  int lane = (idx >> 4) & 31;
  int nt   = (idx >> 9) & (NT_ - 1);
  int kt   = (idx >> 14) & (KTH_ - 1);
  int dir  = idx >> 16;
  int k = kt * 32 + ((lane < 16) ? 0 : 8) + (j & 7) + ((j & 8) << 1);
  int g = nt * 16 + (lane & 15);
  const float* w = dir ? wb : wf;
  out[idx] = (_Float16)w[g * H_ + k];
}

// ---------------------------------------------------------------------------
// Kernel 4: embed+conv+BN1+relu+maxpool via table -> swizzled A-tiles
//   one block per (l, bt); tile staged in LDS, flushed with b128 stores
//   featpack layout: [l][bt][kt][lane][16]  (K = c*5+p, zero-padded to 224)
// ---------------------------------------------------------------------------
__global__ void feat_kernel(const int* __restrict__ x,
                            const float* __restrict__ Mfold,
                            const float* __restrict__ biasf,
                            _Float16* __restrict__ featpack) {
  __shared__ float tab[3 * A_ * C_];
  __shared__ float bf[C_];
  __shared__ __align__(16) _Float16 tile[KTF_ * 32 * 16];  // 3584 halves = 7 KB
  __shared__ int chs[16][8];
  int tid = threadIdx.x;
  int blk = blockIdx.x;                 // l*BT_ + bt
  int l = blk / BT_, bt = blk % BT_;

  for (int i = tid; i < 3 * A_ * C_; i += 256) tab[i] = Mfold[i];
  if (tid < C_) bf[tid] = biasf[tid];
  for (int i = tid; i < KTF_ * 32 * 16; i += 256) tile[i] = (_Float16)0.f;
  if (tid < 16 * W_) {
    int r = tid / W_, p = tid % W_;
    chs[r][p] = x[(((bt * 16 + r) * L_) + l) * W_ + p];
  }
  __syncthreads();

  for (int i = tid; i < 16 * C_; i += 256) {     // 640 (row, channel) items
    int row = i & 15, c = i >> 4;
    float yr[W_];
    for (int p = 0; p < W_; ++p) {
      float v = bf[c];
      if (p > 0)      v += tab[(0 * A_ + chs[row][p - 1]) * C_ + c];
                      v += tab[(1 * A_ + chs[row][p    ]) * C_ + c];
      if (p < W_ - 1) v += tab[(2 * A_ + chs[row][p + 1]) * C_ + c];
      yr[p] = fmaxf(v, 0.f);
    }
    for (int p = 0; p < W_; ++p) {
      float pv = yr[p];
      if (p > 0)      pv = fmaxf(pv, yr[p - 1]);
      if (p < W_ - 1) pv = fmaxf(pv, yr[p + 1]);
      int K = c * W_ + p;
      int kt = K >> 5, ko = K & 31, half, j;
      a_swizzle(ko, half, j);
      tile[(kt * 32 + (half * 16 + row)) * 16 + j] = (_Float16)pv;
    }
  }
  __syncthreads();

  // coalesced flush: 448 x int4
  int4* gp = (int4*)(featpack + ((size_t)blk * KTF_) * 512);
  const int4* sp = (const int4*)tile;
  for (int i = tid; i < KTF_ * 32 * 16 / 8; i += 256) gp[i] = sp[i];
}

// ---------------------------------------------------------------------------
// Kernel 5: fused LSTM step:  z = feat*W_ih^T + h_prev*W_hh^T + b ; gates ; h,c
//   one block (8 waves) per 16-row tile; A-tiles register-resident, two
//   interleaved accumulator chains with software-pipelined B-tile loads.
// ---------------------------------------------------------------------------
__global__ void __launch_bounds__(256) lstm_step(
    const _Float16* __restrict__ featpack_l,
    const _Float16* __restrict__ wihp,
    const _Float16* __restrict__ whhp,
    const float*    __restrict__ bias,
    const _Float16* __restrict__ hpack_in,
    _Float16*       __restrict__ hpack_out,
    float*          __restrict__ cstate,
    _Float16*       __restrict__ hs_out,
    int l, int first) {
  __shared__ float zl[16][G4_];
  int tid = threadIdx.x;
  int wave = tid >> 5, lane = tid & 31;
  int bt = blockIdx.x;

  const _Float16* fpb = featpack_l + (size_t)bt * KTF_ * 512;
  const _Float16* hpb = hpack_in   + (size_t)bt * KTH_ * 512;

  // hoist all A-tiles (invariant across the wave's 4 N-tiles)
  v16h afeat[KTF_];
  #pragma unroll
  for (int kt = 0; kt < KTF_; ++kt)
    afeat[kt] = *(const v16h*)(fpb + ((size_t)kt * 32 + lane) * 16);
  v16h ah[KTH_] = {};
  if (!first) {
    #pragma unroll
    for (int kt = 0; kt < KTH_; ++kt)
      ah[kt] = *(const v16h*)(hpb + ((size_t)kt * 32 + lane) * 16);
  }

  for (int qp = 0; qp < 2; ++qp) {
    int nt0 = wave * 4 + qp * 2;
    // per-lane bases for the two adjacent N-tiles (nt stride = 512 halves)
    const _Float16* bih = wihp + (size_t)nt0 * 512 + (size_t)lane * 16;
    const _Float16* bhh = whhp + (size_t)nt0 * 512 + (size_t)lane * 16;
    v8f acc0 = {}, acc1 = {};
    // software pipeline: b-pair for kt prefetched one WMMA-pair ahead
    v16h b0 = *(const v16h*)(bih);
    v16h b1 = *(const v16h*)(bih + 512);
    #pragma unroll
    for (int kt = 0; kt < KTF_; ++kt) {
      v16h nb0 = {}, nb1 = {};
      if (kt + 1 < KTF_) {
        nb0 = *(const v16h*)(bih + (size_t)(kt + 1) * NT_ * 512);
        nb1 = *(const v16h*)(bih + (size_t)(kt + 1) * NT_ * 512 + 512);
      } else if (!first) {                         // bridge into recurrent GEMM
        nb0 = *(const v16h*)(bhh);
        nb1 = *(const v16h*)(bhh + 512);
      }
      acc0 = __builtin_amdgcn_wmma_f32_16x16x32_f16(false, afeat[kt], false, b0,
                                                    (short)0, acc0, false, false);
      acc1 = __builtin_amdgcn_wmma_f32_16x16x32_f16(false, afeat[kt], false, b1,
                                                    (short)0, acc1, false, false);
      b0 = nb0; b1 = nb1;
    }
    if (!first) {
      #pragma unroll
      for (int kt = 0; kt < KTH_; ++kt) {
        v16h nb0 = {}, nb1 = {};
        if (kt + 1 < KTH_) {
          nb0 = *(const v16h*)(bhh + (size_t)(kt + 1) * NT_ * 512);
          nb1 = *(const v16h*)(bhh + (size_t)(kt + 1) * NT_ * 512 + 512);
        }
        acc0 = __builtin_amdgcn_wmma_f32_16x16x32_f16(false, ah[kt], false, b0,
                                                      (short)0, acc0, false, false);
        acc1 = __builtin_amdgcn_wmma_f32_16x16x32_f16(false, ah[kt], false, b1,
                                                      (short)0, acc1, false, false);
        b0 = nb0; b1 = nb1;
      }
    }
    // D layout: VGPR r, lane -> row = r + 8*(lane>>4), col = lane&15
    int n = lane & 15, half = lane >> 4;
    #pragma unroll
    for (int r = 0; r < 8; ++r) {
      zl[r + half * 8][nt0 * 16 + n]        = acc0[r];
      zl[r + half * 8][(nt0 + 1) * 16 + n]  = acc1[r];
    }
  }
  __syncthreads();

  // gate phase: each thread owns 8 CONSECUTIVE units of one row so that all
  // global accesses (bias, cstate, hs, hpack) are contiguous 16/32-byte runs.
  {
    int row = tid >> 4;
    int u0 = (tid & 15) * 8;
    int bg = bt * 16 + row;
    float cl[8];
    _Float16 hl[8];
    const float* csp = cstate + (size_t)bg * H_ + u0;
    #pragma unroll
    for (int e = 0; e < 8; ++e) {
      int u = u0 + e;
      float zi = zl[row][u          ] + bias[u          ];
      float zf = zl[row][u +     H_ ] + bias[u +     H_ ];
      float zg = zl[row][u + 2 * H_ ] + bias[u + 2 * H_ ];
      float zo = zl[row][u + 3 * H_ ] + bias[u + 3 * H_ ];
      float cold = first ? 0.f : csp[e];
      float si = 1.f / (1.f + __expf(-zi));
      float sf = 1.f / (1.f + __expf(-zf));
      float so = 1.f / (1.f + __expf(-zo));
      float tg = tanhf(zg);
      float c  = sf * cold + si * tg;
      cl[e] = c;
      hl[e] = (_Float16)(so * tanhf(c));
    }
    float* csw = cstate + (size_t)bg * H_ + u0;
    #pragma unroll
    for (int e = 0; e < 8; ++e) csw[e] = cl[e];
    _Float16* hsw = hs_out + ((size_t)bg * L_ + l) * H_ + u0;
    #pragma unroll
    for (int e = 0; e < 8; ++e) hsw[e] = hl[e];
    // the aligned 8-unit run maps to one contiguous 8-half run in A-layout:
    int kt = u0 >> 5, ko = u0 & 31;
    int half = (ko >> 3) & 1;
    int j0 = (ko & 16) >> 1;                       // 0 or 8
    _Float16* hpw = hpack_out +
        (((size_t)bt * KTH_ + kt) * 32 + (half * 16 + row)) * 16 + j0;
    #pragma unroll
    for (int e = 0; e < 8; ++e) hpw[e] = hl[e];
  }
}

// ---------------------------------------------------------------------------
// Kernel 6: BN2 + FC -> emissions [N,3]
// ---------------------------------------------------------------------------
__global__ void emis_kernel(const _Float16* __restrict__ hs,
                            const float* __restrict__ g2, const float* __restrict__ b2,
                            const float* __restrict__ m2, const float* __restrict__ v2,
                            const float* __restrict__ fcw, const float* __restrict__ fcb,
                            float* __restrict__ emis) {
  __shared__ float sc[2 * H_], sb[2 * H_], w0[2 * H_], w1[2 * H_], w2[2 * H_];
  int tid = threadIdx.x;
  if (tid < 2 * H_) {
    float s = g2[tid] * rsqrtf(v2[tid] + EPS_);
    sc[tid] = s;
    sb[tid] = b2[tid] - m2[tid] * s;
    w0[tid] = fcw[0 * 2 * H_ + tid];
    w1[tid] = fcw[1 * 2 * H_ + tid];
    w2[tid] = fcw[2 * 2 * H_ + tid];
  }
  __syncthreads();
  int rg = blockIdx.x * blockDim.x + tid;
  const _Float16* hf = hs + (size_t)rg * H_;
  const _Float16* hb = hs + (size_t)N_ * H_ + (size_t)rg * H_;
  float e0 = fcb[0], e1 = fcb[1], e2 = fcb[2];
  for (int c = 0; c < H_; ++c) {
    float hn = (float)hf[c] * sc[c] + sb[c];
    e0 += w0[c] * hn; e1 += w1[c] * hn; e2 += w2[c] * hn;
  }
  for (int c = 0; c < H_; ++c) {
    int cc = H_ + c;
    float hn = (float)hb[c] * sc[cc] + sb[cc];
    e0 += w0[cc] * hn; e1 += w1[cc] * hn; e2 += w2[cc] * hn;
  }
  emis[(size_t)rg * 3 + 0] = e0;
  emis[(size_t)rg * 3 + 1] = e1;
  emis[(size_t)rg * 3 + 2] = e2;
}

// ---------------------------------------------------------------------------
// Kernel 7: Viterbi decode, one thread per sequence (T=3)
//   out[0..B) = best path score ; out[B + b*L + l] = tag (as float)
// ---------------------------------------------------------------------------
__global__ void viterbi_kernel(const float* __restrict__ emis,
                               const int* __restrict__ x,
                               const float* __restrict__ trans,
                               const float* __restrict__ start_t,
                               const float* __restrict__ end_t,
                               float* __restrict__ out) {
  int b = blockIdx.x * blockDim.x + threadIdx.x;
  if (b >= B_) return;
  const float* em = emis + (size_t)b * L_ * T_;
  float tr[T_ * T_];
  for (int i = 0; i < T_ * T_; ++i) tr[i] = trans[i];
  float score[T_];
  for (int t = 0; t < T_; ++t) score[t] = start_t[t] + em[t];
  signed char hist[L_ - 1][T_];
  for (int l = 1; l < L_; ++l) {
    bool msk = x[(b * L_ + l) * W_ + 2] > 0;
    float ns[T_];
    for (int tn = 0; tn < T_; ++tn) {
      float best = -3.4e38f; int arg = 0;
      for (int tp = 0; tp < T_; ++tp) {
        float cnd = score[tp] + tr[tp * T_ + tn];
        if (cnd > best) { best = cnd; arg = tp; }     // first-max tie-break
      }
      hist[l - 1][tn] = (signed char)arg;
      ns[tn] = best + em[l * T_ + tn];
    }
    if (msk) for (int t = 0; t < T_; ++t) score[t] = ns[t];
  }
  float best = -3.4e38f; int last = 0;
  for (int t = 0; t < T_; ++t) {
    float s = score[t] + end_t[t];
    if (s > best) { best = s; last = t; }
  }
  int tags[L_]; tags[L_ - 1] = last;
  int tag = last;
  for (int l = L_ - 2; l >= 0; --l) {
    int prev = hist[l][tag];
    if (x[(b * L_ + (l + 1)) * W_ + 2] > 0) tag = prev;
    tags[l] = tag;
  }
  out[b] = best;
  for (int l = 0; l < L_; ++l) out[B_ + (size_t)b * L_ + l] = (float)tags[l];
}

// ---------------------------------------------------------------------------
extern "C" void kernel_launch(void* const* d_in, const int* in_sizes, int n_in,
                              void* d_out, int out_size, void* d_ws, size_t ws_size,
                              hipStream_t stream) {
  (void)in_sizes; (void)n_in; (void)out_size; (void)ws_size;
  const int*   x       = (const int*)  d_in[0];
  const float* emb     = (const float*)d_in[1];
  const float* conv_w  = (const float*)d_in[2];
  const float* conv_b  = (const float*)d_in[3];
  const float* bn1_g   = (const float*)d_in[4];
  const float* bn1_b   = (const float*)d_in[5];
  const float* bn1_m   = (const float*)d_in[6];
  const float* bn1_v   = (const float*)d_in[7];
  const float* w_ih_f  = (const float*)d_in[8];
  const float* w_hh_f  = (const float*)d_in[9];
  const float* b_f     = (const float*)d_in[10];
  const float* w_ih_b  = (const float*)d_in[11];
  const float* w_hh_b  = (const float*)d_in[12];
  const float* b_b     = (const float*)d_in[13];
  const float* bn2_g   = (const float*)d_in[14];
  const float* bn2_b   = (const float*)d_in[15];
  const float* bn2_m   = (const float*)d_in[16];
  const float* bn2_v   = (const float*)d_in[17];
  const float* fc_w    = (const float*)d_in[18];
  const float* fc_b    = (const float*)d_in[19];
  const float* trans   = (const float*)d_in[20];
  const float* start_t = (const float*)d_in[21];
  const float* end_t   = (const float*)d_in[22];

  char* ws = (char*)d_ws;
  size_t off = 0;
  auto take = [&](size_t bytes) -> char* {
    char* p = ws + off;
    off = (off + bytes + 255) & ~(size_t)255;
    return p;
  };
  float*    Mfold  = (float*)   take((size_t)3 * A_ * C_ * 4);          // 14.4 KB
  float*    biasf  = (float*)   take((size_t)C_ * 4);
  _Float16* wihp   = (_Float16*)take((size_t)2 * KTF_ * NT_ * 512 * 2); // 448 KB
  _Float16* whhp   = (_Float16*)take((size_t)2 * KTH_ * NT_ * 512 * 2); // 256 KB
  _Float16* featp  = (_Float16*)take((size_t)L_ * BT_ * KTF_ * 512 * 2);// 55 MB
  _Float16* hs     = (_Float16*)take((size_t)2 * N_ * H_ * 2);          // 63 MB
  _Float16* hpA    = (_Float16*)take((size_t)BT_ * KTH_ * 512 * 2);     // 1 MB
  _Float16* hpB    = (_Float16*)take((size_t)BT_ * KTH_ * 512 * 2);     // 1 MB
  float*    cstate = (float*)   take((size_t)B_ * H_ * 4);              // 2 MB
  float*    emis   = (float*)   take((size_t)N_ * T_ * 4);              // 1.5 MB

  build_tables<<<15, 256, 0, stream>>>(emb, conv_w, conv_b, bn1_g, bn1_b, bn1_m, bn1_v,
                                       Mfold, biasf);
  pack_wih<<<896, 256, 0, stream>>>(w_ih_f, w_ih_b, wihp);
  pack_whh<<<512, 256, 0, stream>>>(w_hh_f, w_hh_b, whhp);
  feat_kernel<<<L_ * BT_, 256, 0, stream>>>(x, Mfold, biasf, featp);

  for (int dir = 0; dir < 2; ++dir) {
    const _Float16* wih_d = wihp + (size_t)dir * KTF_ * NT_ * 512;
    const _Float16* whh_d = whhp + (size_t)dir * KTH_ * NT_ * 512;
    const float* bias_d   = dir ? b_b : b_f;
    _Float16* hs_d        = hs + (size_t)dir * N_ * H_;
    for (int step = 0; step < L_; ++step) {
      int l = dir ? (L_ - 1 - step) : step;
      _Float16* hin  = (step & 1) ? hpB : hpA;
      _Float16* hout = (step & 1) ? hpA : hpB;
      lstm_step<<<BT_, 256, 0, stream>>>(featp + (size_t)l * BT_ * KTF_ * 512,
                                         wih_d, whh_d, bias_d, hin, hout,
                                         cstate, hs_d, l, step == 0 ? 1 : 0);
    }
  }

  emis_kernel<<<N_ / 256, 256, 0, stream>>>(hs, bn2_g, bn2_b, bn2_m, bn2_v,
                                            fc_w, fc_b, emis);
  viterbi_kernel<<<B_ / 256, 256, 0, stream>>>(emis, x, trans, start_t, end_t,
                                               (float*)d_out);
}
